// QLinear_58823872086321
// MI455X (gfx1250) — 8-bit matrix operand order repaired and checked by reading, compiled, not run on hardware
//
#include <hip/hip_runtime.h>
#include <stdint.h>

typedef __attribute__((ext_vector_type(8))) int       v8i;
typedef __attribute__((ext_vector_type(4))) int       v4i;
typedef __attribute__((ext_vector_type(8))) _Float16  h8;

#define DIN   4096
#define DOUT  16384
#define M_TOT 1024
#define BK    64
#define NSTAGE (DIN / BK)
#define LDPITCH 80

#if __has_builtin(__builtin_amdgcn_global_load_async_to_lds_b128)
#define HAVE_ASYNC_LDS 1
#else
#define HAVE_ASYNC_LDS 0
#endif

#if HAVE_ASYNC_LDS
typedef __attribute__((address_space(1))) v4i gv4i;
typedef __attribute__((address_space(3))) v4i lv4i;

static __device__ __forceinline__ void async_cp16(const int8_t* g, int8_t* l) {
  __builtin_amdgcn_global_load_async_to_lds_b128((gv4i*)g, (lv4i*)l, 0, 0);
}

#if __has_builtin(__builtin_amdgcn_s_wait_asynccnt)
#define WAIT_ASYNC(N) __builtin_amdgcn_s_wait_asynccnt(N)
#else
#define WAIT_ASYNC(N) asm volatile("s_wait_asynccnt " #N ::: "memory")
#endif
#endif

__global__ void __launch_bounds__(256)
quant_rows(const _Float16* __restrict__ x,
           int8_t* __restrict__ xq,
           float* __restrict__ rowmax) {
  const int row = blockIdx.x;
  const int t   = threadIdx.x;

  const h8* xv = (const h8*)(x + (size_t)row * DIN);
  h8 c0 = xv[2 * t + 0];
  h8 c1 = xv[2 * t + 1];

  float v[16];
  float m = 0.0f;
#pragma unroll
  for (int i = 0; i < 8; ++i) { v[i] = (float)c0[i]; v[8 + i] = (float)c1[i]; }
#pragma unroll
  for (int i = 0; i < 16; ++i) m = fmaxf(m, fabsf(v[i]));

  __shared__ float red[256];
  red[t] = m;
  __syncthreads();
#pragma unroll
  for (int s = 128; s > 0; s >>= 1) {
    if (t < s) red[t] = fmaxf(red[t], red[t + s]);
    __syncthreads();
  }
  const float amax = fmaxf(red[0], 1e-5f);
  if (t == 0) rowmax[row] = amax;

  const float scale = 127.0f / amax;
  uint32_t w[4];
#pragma unroll
  for (int j = 0; j < 4; ++j) {
    uint32_t p = 0;
#pragma unroll
    for (int i = 0; i < 4; ++i) {
      int q = (int)__builtin_rintf(v[4 * j + i] * scale);
      p |= ((uint32_t)(uint8_t)(int8_t)q) << (8 * i);
    }
    w[j] = p;
  }
  v4i pk = { (int)w[0], (int)w[1], (int)w[2], (int)w[3] };
  *(v4i*)(xq + (size_t)row * DIN + t * 16) = pk;
}

static __device__ __forceinline__ void
compute_slab(const int8_t* __restrict__ pA, const int8_t* __restrict__ pB,
             v8i acc[2][4]) {
  v8i af[2];
#pragma unroll
  for (int mt = 0; mt < 2; ++mt) {
    const int* q = (const int*)(pA + mt * (16 * LDPITCH));
    v8i a;
    a[0] = q[0];  a[1] = q[1];
    a[2] = q[2];  a[3] = q[3];
    a[4] = q[8];  a[5] = q[9];
    a[6] = q[10]; a[7] = q[11];
    af[mt] = a;
  }
  v8i bf[4];
#pragma unroll
  for (int nt = 0; nt < 4; ++nt) {
    const v4i* q = (const v4i*)(pB + nt * (16 * LDPITCH));
    bf[nt] = __builtin_shufflevector(q[0], q[2], 0, 1, 2, 3, 4, 5, 6, 7);
  }
#pragma unroll
  for (int mt = 0; mt < 2; ++mt)
#pragma unroll
    for (int nt = 0; nt < 4; ++nt)
      acc[mt][nt] = __builtin_amdgcn_wmma_i32_16x16x64_iu8(
           true, af[mt],  true, bf[nt],
          acc[mt][nt],  false,  false);
}

#if !HAVE_ASYNC_LDS
struct Stage4 { v4i a0, a1, b0, b1; };
#endif

__global__ void __launch_bounds__(256)
int8_gemm_wmma(const int8_t* __restrict__ xq,
               const int8_t* __restrict__ CB,
               const float* __restrict__ rowmax,
               const float* __restrict__ SCB,
               _Float16* __restrict__ out) {
#if HAVE_ASYNC_LDS
  __shared__ int8_t lA[4][128 * LDPITCH];
  __shared__ int8_t lB[4][128 * LDPITCH];
#else
  __shared__ int8_t lA[2][128 * LDPITCH];
  __shared__ int8_t lB[2][128 * LDPITCH];
#endif

  const int n0 = blockIdx.x * 128;
  const int m0 = blockIdx.y * 128;
  const int t      = threadIdx.x;
  const int lane   = t & 31;
  const int wave   = t >> 5;
  const int wm     = wave >> 1;
  const int wn     = wave & 1;
  const int lane16 = lane & 15;
  const int laneHi = lane >> 4;

  const int r0  = t >> 2;
  const int k0c = (t & 3) * 16;
  const int r1  = (t + 256) >> 2;

  const int8_t* gA = xq + (size_t)m0 * DIN;
  const int8_t* gB = CB + (size_t)n0 * DIN;

  const int aBase = (wm * 32 + lane16) * LDPITCH + laneHi * 16;
  const int bBase = (wn * 64 + lane16) * LDPITCH + laneHi * 16;

  const int lo0 = r0 * LDPITCH + k0c;
  const int lo1 = r1 * LDPITCH + k0c;
  const size_t go0 = (size_t)r0 * DIN + k0c;
  const size_t go1 = (size_t)r1 * DIN + k0c;

  v8i acc[2][4];
#pragma unroll
  for (int i = 0; i < 2; ++i)
#pragma unroll
    for (int j = 0; j < 4; ++j)
      acc[i][j] = (v8i){0, 0, 0, 0, 0, 0, 0, 0};

#if HAVE_ASYNC_LDS
  int8_t* const A0 = lA[0]; int8_t* const A1 = lA[1];
  int8_t* const A2 = lA[2]; int8_t* const A3 = lA[3];
  int8_t* const B0 = lB[0]; int8_t* const B1 = lB[1];
  int8_t* const B2 = lB[2]; int8_t* const B3 = lB[3];

  const int8_t* pA0 = gA + go0;
  const int8_t* pA1 = gA + go1;
  const int8_t* pB0 = gB + go0;
  const int8_t* pB1 = gB + go1;

#define PREFETCH_OFF(OFF, BA, BB)                  \
  do {                                             \
    async_cp16(pA0 + (OFF), (BA) + lo0);           \
    async_cp16(pA1 + (OFF), (BA) + lo1);           \
    async_cp16(pB0 + (OFF), (BB) + lo0);           \
    async_cp16(pB1 + (OFF), (BB) + lo1);           \
  } while (0)

#define STAGE_PF(CURA, CURB, PFOFF, PFA, PFB)          \
  do {                                                 \
    PREFETCH_OFF((PFOFF), (PFA), (PFB));               \
    compute_slab((CURA) + aBase, (CURB) + bBase, acc); \
    WAIT_ASYNC(4);                                     \
    __syncthreads();                                   \
  } while (0)

  PREFETCH_OFF(0,  A0, B0);
  PREFETCH_OFF(64, A1, B1);
  WAIT_ASYNC(4);
  __syncthreads();

#pragma unroll 1
  for (int q = 0; q < NSTAGE / 4 - 1; ++q) {
    STAGE_PF(A0, B0, 128, A2, B2);
    STAGE_PF(A1, B1, 192, A3, B3);
    STAGE_PF(A2, B2, 256, A0, B0);
    STAGE_PF(A3, B3, 320, A1, B1);
    pA0 += 4 * BK; pA1 += 4 * BK; pB0 += 4 * BK; pB1 += 4 * BK;
  }
  STAGE_PF(A0, B0, 128, A2, B2);
  STAGE_PF(A1, B1, 192, A3, B3);
  compute_slab(A2 + aBase, B2 + bBase, acc);
  WAIT_ASYNC(0);
  __syncthreads();
  compute_slab(A3 + aBase, B3 + bBase, acc);

#undef STAGE_PF
#undef PREFETCH_OFF

#else
#define PIPE_STAGE(CA, CB_, NA, NB, GK)                          \
  do {                                                           \
    Stage4 st;                                                   \
    st.a0 = *(const v4i*)(gA + (GK) + go0);                      \
    st.a1 = *(const v4i*)(gA + (GK) + go1);                      \
    st.b0 = *(const v4i*)(gB + (GK) + go0);                      \
    st.b1 = *(const v4i*)(gB + (GK) + go1);                      \
    compute_slab((CA) + aBase, (CB_) + bBase, acc);              \
    *(v4i*)((NA) + lo0) = st.a0;                                 \
    *(v4i*)((NA) + lo1) = st.a1;                                 \
    *(v4i*)((NB) + lo0) = st.b0;                                 \
    *(v4i*)((NB) + lo1) = st.b1;                                 \
    __syncthreads();                                             \
  } while (0)

  {
    v4i a0 = *(const v4i*)(gA + go0);
    v4i a1 = *(const v4i*)(gA + go1);
    v4i b0 = *(const v4i*)(gB + go0);
    v4i b1 = *(const v4i*)(gB + go1);
    *(v4i*)(lA[0] + lo0) = a0;
    *(v4i*)(lA[0] + lo1) = a1;
    *(v4i*)(lB[0] + lo0) = b0;
    *(v4i*)(lB[0] + lo1) = b1;
  }
  __syncthreads();

  for (int p = 0; p < NSTAGE / 2 - 1; ++p) {
    PIPE_STAGE(lA[0], lB[0], lA[1], lB[1], (size_t)(2 * p + 1) * BK);
    PIPE_STAGE(lA[1], lB[1], lA[0], lB[0], (size_t)(2 * p + 2) * BK);
  }
  PIPE_STAGE(lA[0], lB[0], lA[1], lB[1], (size_t)(NSTAGE - 1) * BK);
  compute_slab(lA[1] + aBase, lB[1] + bBase, acc);
#undef PIPE_STAGE
#endif

  const float inv = 1.0f / (127.0f * 127.0f);
#pragma unroll
  for (int mt = 0; mt < 2; ++mt) {
    float am[8];
#pragma unroll
    for (int j = 0; j < 8; ++j)
      am[j] = rowmax[m0 + wm * 32 + mt * 16 + laneHi * 8 + j] * inv;
#pragma unroll
    for (int nt = 0; nt < 4; ++nt) {
      const int n = n0 + wn * 64 + nt * 16 + lane16;
      const float sc = SCB[n];
#pragma unroll
      for (int j = 0; j < 8; ++j) {
        const int m = m0 + wm * 32 + mt * 16 + laneHi * 8 + j;
        out[(size_t)m * DOUT + n] =
            (_Float16)((float)acc[mt][nt][j] * am[j] * sc);
      }
    }
  }
}

extern "C" void kernel_launch(void* const* d_in, const int* in_sizes, int n_in,
                              void* d_out, int out_size, void* d_ws,
                              size_t ws_size, hipStream_t stream) {
  const _Float16* x  = (const _Float16*)d_in[0];
  const int8_t* CB   = (const int8_t*)d_in[1];
  const float* SCB   = (const float*)d_in[2];
  _Float16* out      = (_Float16*)d_out;

  int8_t* xq    = (int8_t*)d_ws;
  float* rowmax = (float*)((char*)d_ws + (size_t)M_TOT * DIN);

  quant_rows<<<M_TOT, 256, 0, stream>>>(x, xq, rowmax);

  dim3 grid(DOUT / 128, M_TOT / 128);
  int8_gemm_wmma<<<grid, 256, 0, stream>>>(xq, CB, rowmax, SCB, out);
}
